// Correlation_16020228014580
// MI455X (gfx1250) — matrix operand order repaired and checked by reading, not compiled, not run on hardware
//
#include <hip/hip_runtime.h>

// Correlation via banded-Gram WMMA (gfx1250, wave32).
// B=16, C=256, H=64, W=96, patch 21, dilation 2.
#define B_   16
#define C_   256
#define H_   64
#define W_   96
#define P_   21
#define SPAR 48          // positions per parity class (W/2)
#define RS1  264         // sA row stride in f16 elems (256 + 8 pad -> 528B, 16B-aligned, conflict-free)
#define RS2  40          // sB row stride in f16 elems (32 + 8 pad -> 80B)
#define KC   32          // channel chunk (one K=32 WMMA per term)
#define NCHUNK (C_/KC)

typedef __attribute__((ext_vector_type(16))) _Float16 v16h;
typedef __attribute__((ext_vector_type(8)))  _Float16 v8h;
typedef __attribute__((ext_vector_type(8)))  float    v8f;

static __device__ inline v16h mk16(const _Float16* __restrict__ p0,
                                   const _Float16* __restrict__ p1) {
  v8h a = *(const v8h*)p0;   // ds_load_b128
  v8h b = *(const v8h*)p1;   // ds_load_b128
  v16h r;
#pragma unroll
  for (int i = 0; i < 8; ++i) { r[i] = a[i]; r[i + 8] = b[i]; }
  return r;
}

__global__ __launch_bounds__(192)
void corr_wmma_kernel(const float* __restrict__ in1,
                      const float* __restrict__ in2,
                      float* __restrict__ out) {
  // [hi/lo][parity][pos][c] transposed layouts so A and B fragments are b128 LDS loads
  __shared__ __align__(16) _Float16 sA[2][2][SPAR * RS1]; // in1 row, full C resident (~101KB)
  __shared__ __align__(16) _Float16 sB[2][2][80 * RS2];   // in2 chunk, t in [-16,64) halo (~26KB)

  const int tid  = threadIdx.x;
  const int lane = tid & 31;
  const int wid  = tid >> 5;          // 6 waves
  const int parw = wid & 1;           // parity class this wave computes
  const int s0   = (wid >> 1) * 16;   // s-tile base (0,16,32)

  const int b = blockIdx.x / H_;
  const int y = blockIdx.x % H_;

  // ---- phase 0: stage in1[b,:,y,:] as f16 hi/lo, parity-split, [s][c] ----
  {
    const int x   = tid % W_;
    const int cb  = tid / W_;          // 0..1
    const int par = x & 1, s = x >> 1;
    const float* p = in1 + (((size_t)b * C_) * H_ + y) * W_ + x;
#pragma unroll 4
    for (int c = cb; c < C_; c += 2) {
      float v = p[(size_t)c * H_ * W_];
      _Float16 h = (_Float16)v;
      _Float16 l = (_Float16)(v - (float)h);
      sA[0][par][s * RS1 + c] = h;
      sA[1][par][s * RS1 + c] = l;
    }
  }
  // zero sB halo rows (t<0, t>=48); never overwritten afterwards -> x-edge padding for free
  for (int i = tid; i < 2 * 2 * 32 * KC; i += 192) {
    int pl  = i >> 11;
    int par = (i >> 10) & 1;
    int rr  = (i >> 5) & 31;
    int c   = i & 31;
    int r   = (rr < 16) ? rr : (rr + 48);
    sB[pl][par][r * RS2 + c] = (_Float16)0;
  }
  __syncthreads();

  for (int pi = 0; pi < P_; ++pi) {
    const int ysrc = y + (pi - 10) * 2;          // uniform across block
    if (ysrc < 0 || ysrc >= H_) {                // y-padding: whole pi plane is zero
      for (int i = tid; i < P_ * W_; i += 192) {
        int pj = i / W_, xx = i % W_;
        out[((((size_t)b * P_ + pi) * P_ + pj) * H_ + y) * W_ + xx] = 0.0f;
      }
      continue;
    }

    const v8f vz = {0.f, 0.f, 0.f, 0.f, 0.f, 0.f, 0.f, 0.f};
    v8f acc[3]; acc[0] = vz; acc[1] = vz; acc[2] = vz;   // 3 t-tiles of the band

    for (int k = 0; k < NCHUNK; ++k) {
      __syncthreads();               // protect sB readers from previous chunk
      { // stage in2[b, c0:c0+32, ysrc, :] (hi/lo, parity-split, [t+16][c])
        const int x   = tid % W_;
        const int cb  = tid / W_;
        const int par = x & 1, s = x >> 1;
        const float* p = in2 + (((size_t)(b * C_ + k * KC)) * H_ + ysrc) * W_ + x;
#pragma unroll 4
        for (int c = cb; c < KC; c += 2) {
          float v = p[(size_t)c * H_ * W_];
          _Float16 h = (_Float16)v;
          _Float16 l = (_Float16)(v - (float)h);
          sB[0][par][(s + 16) * RS2 + c] = h;
          sB[1][par][(s + 16) * RS2 + c] = l;
        }
      }
      __syncthreads();

      // A fragment: M = lane%16 -> row s0+M; K packing per ISA 16-bit A 16x32 layout
      const int row = s0 + (lane & 15);
      const int ch8 = (lane >> 4) * 8;
      const int c0  = k * KC;
      const _Float16* aHp = &sA[0][parw][row * RS1 + c0];
      const _Float16* aLp = &sA[1][parw][row * RS1 + c0];
      v16h a_hi = mk16(aHp + ch8, aHp + 16 + ch8);
      v16h a_lo = mk16(aLp + ch8, aLp + 16 + ch8);

      const int bc = (lane >> 4) * 8;    // B: K = (lane/16)*8 + e (e<8), 16 + (lane/16)*8 + (e-8) (e>=8), N = lane%16
#pragma unroll
      for (int tt = 0; tt < 3; ++tt) {
        const int br = s0 + 16 * tt + (lane & 15);  // LDS row = t + 16
        const _Float16* bHp = &sB[0][parw][br * RS2 + bc];
        const _Float16* bLp = &sB[1][parw][br * RS2 + bc];
        v16h b_hi = mk16(bHp, bHp + 16);
        v16h b_lo = mk16(bLp, bLp + 16);
        // fp32-accurate product via f16 hi/lo split: hi*hi + lo*hi + hi*lo
        acc[tt] = __builtin_amdgcn_wmma_f32_16x16x32_f16(false, a_hi, false, b_hi,
                                                         (short)0, acc[tt], false, false);
        acc[tt] = __builtin_amdgcn_wmma_f32_16x16x32_f16(false, a_lo, false, b_hi,
                                                         (short)0, acc[tt], false, false);
        acc[tt] = __builtin_amdgcn_wmma_f32_16x16x32_f16(false, a_hi, false, b_lo,
                                                         (short)0, acc[tt], false, false);
      }
    }

    // epilogue: scatter the 21-lag band. C/D layout: M = vgpr + 8*(lane>=16), N = lane%16
    {
      const int N  = lane & 15;
      const int mb = (lane >> 4) * 8;
#pragma unroll
      for (int tt = 0; tt < 3; ++tt) {
#pragma unroll
        for (int i = 0; i < 8; ++i) {
          int s   = s0 + mb + i;
          int t   = s0 + 16 * (tt - 1) + N;
          int lag = t - s;                 // even-x-displacement lag within parity class
          if (lag >= -10 && lag <= 10) {
            int pj = lag + 10;
            int xx = 2 * s + parw;
            out[((((size_t)b * P_ + pi) * P_ + pj) * H_ + y) * W_ + xx] = acc[tt][i];
          }
        }
      }
    }
  }
}

extern "C" void kernel_launch(void* const* d_in, const int* in_sizes, int n_in,
                              void* d_out, int out_size, void* d_ws, size_t ws_size,
                              hipStream_t stream) {
  (void)in_sizes; (void)n_in; (void)d_ws; (void)ws_size; (void)out_size;
  const float* in1 = (const float*)d_in[0];
  const float* in2 = (const float*)d_in[1];
  float* out = (float*)d_out;
  dim3 grid(B_ * H_);   // one workgroup per (b, y)
  dim3 block(192);      // 6 wave32 = (2 parities x 3 s-tiles)
  hipLaunchKernelGGL(corr_wmma_kernel, grid, block, 0, stream, in1, in2, out);
}
